// Gauss_10831907520641
// MI455X (gfx1250) — matrix operand order repaired and checked by reading, not compiled, not run on hardware
//
#include <hip/hip_runtime.h>

// ---------------------------------------------------------------------------
// Diagonal-Gaussian log-likelihood == one bf16 GEMM (K=2048) + per-m bias.
//   out[n,m] = sum_d x^2[n,d]*(-0.5*prec2[m,d]) + sum_d x[n,d]*(mu*prec2)[m,d]
//              + bias[m]
// MI455X reasoning: 134 GFLOP vs ~200MB of HBM traffic (9us @ 23.3 TB/s).
// f32 WMMA (16x16x4) would be compute-bound; bf16 WMMA (16x16x32, f32 accum)
// lands at the HBM floor with ~1e-4 relative error. x (64MB) + W (8MB) fit in
// the 192MB L2 and are re-read across the grid -> keep them temporal; out
// (128MB, write-once) is stored non-temporal. LDS is ping-pong double
// buffered so next tile's global->LDS staging overlaps the WMMA stream.
// ---------------------------------------------------------------------------

typedef __attribute__((ext_vector_type(16))) __bf16       v16bf;
typedef __attribute__((ext_vector_type(8)))  float        v8f;
typedef __attribute__((ext_vector_type(4)))  __bf16       v4bf;
typedef __attribute__((ext_vector_type(4)))  unsigned int v4u;
typedef __attribute__((ext_vector_type(4)))  float        v4f;

#define NUM_BPE 2000
#define MPAD    2048
#define FEA     1024
#define NTOT    16384   // 16 * 1024 frames

#define BN  128          // block tile rows (n)
#define BM  64           // block tile cols (m)
#define BK  64           // K step (per half: x^2-half and x-half share it)
#define LDA 72           // padded LDS row stride (bf16 elems) -> bank spread
#define ATILE (BN * LDA)
#define BTILE (BM * LDA)

union FragU { v16bf v; v4u q[2]; };   // 32B: one WMMA A/B operand

// ---------------------------------------------------------------------------
// Kernel 1: build W[2048][2048] bf16 and bias[2048] f32 in workspace.
//   W[m, d]       = bf16(-0.5 / cov[m,d])          (pairs with x^2)
//   W[m, 1024+d]  = bf16(mu[m,d] / cov[m,d])       (pairs with x)
//   bias[m]       = -0.5*(D*log2pi + sum mu^2/cov) - 0.5*sum log(cov)
// Rows m in [2000,2048) zeroed so padded WMMA lanes accumulate 0.
// ---------------------------------------------------------------------------
__global__ __launch_bounds__(256) void gauss_prep(
    const float* __restrict__ mu, const float* __restrict__ cov,
    __bf16* __restrict__ W, float* __restrict__ bias)
{
    const int m   = blockIdx.x;      // 0..2047
    const int tid = threadIdx.x;     // 256 threads
    __shared__ float r1[256];
    __shared__ float r2[256];

    float s1 = 0.f, s2 = 0.f;
    if (m < NUM_BPE) {
        for (int d = tid; d < FEA; d += 256) {
            const float c = cov[(size_t)m * FEA + d];
            const float p = 1.0f / c;
            const float u = mu[(size_t)m * FEA + d];
            W[(size_t)m * (2 * FEA) + d]       = (__bf16)(-0.5f * p);
            W[(size_t)m * (2 * FEA) + FEA + d] = (__bf16)(u * p);
            s1 += u * u * p;
            s2 += logf(c);
        }
    } else {
        for (int d = tid; d < 2 * FEA; d += 256)
            W[(size_t)m * (2 * FEA) + d] = (__bf16)0.f;
    }

    r1[tid] = s1; r2[tid] = s2;
    __syncthreads();
    for (int off = 128; off > 0; off >>= 1) {
        if (tid < off) { r1[tid] += r1[tid + off]; r2[tid] += r2[tid + off]; }
        __syncthreads();
    }
    if (tid == 0) {
        const float LOG2PI = 1.8378770664093453f;
        bias[m] = (m < NUM_BPE)
                ? (-0.5f * ((float)FEA * LOG2PI + r1[0]) - 0.5f * r2[0])
                : 0.f;
    }
}

// ---------------------------------------------------------------------------
// Kernel 2: the GEMM. Grid = (MPAD/BM, NTOT/BN). 256 threads = 8 wave32.
// Waves arranged 4(n) x 2(m); each wave owns a 32x32 output tile
// (2x2 grid of 16x16 f32 accumulators). Double-buffered LDS.
// ---------------------------------------------------------------------------
__global__ __launch_bounds__(256) void gauss_gemm(
    const float*  __restrict__ x,
    const __bf16* __restrict__ W,
    const float*  __restrict__ bias,
    float*        __restrict__ out)
{
    __shared__ alignas(16) __bf16 as1[2 * ATILE];   // bf16(x^2), ping-pong
    __shared__ alignas(16) __bf16 as2[2 * ATILE];   // bf16(x)
    __shared__ alignas(16) __bf16 bs1[2 * BTILE];   // W first half
    __shared__ alignas(16) __bf16 bs2[2 * BTILE];   // W second half

    const int tid  = threadIdx.x;
    const int lane = tid & 31;
    const int wid  = tid >> 5;
    const int wrow = (wid >> 1) * 32;   // wave n-offset: 0/32/64/96
    const int wcol = (wid & 1) * 32;    // wave m-offset: 0/32
    const int lr   = lane & 15;
    const int lh   = lane >> 4;

    const int n0 = blockIdx.y * BN;
    const int m0 = blockIdx.x * BM;

    v8f acc[2][2] = {};

    // global->LDS index split
    const int tr = tid >> 4;            // x tile: row within 16-row pass
    const int tc = (tid & 15) * 4;      // x tile: float column
    const int br = tid >> 3;            // W tile: row within 32-row pass
    const int bc = (tid & 7) * 8;       // W tile: bf16 column

    // ---- stage: global -> LDS for one BK slab (x once -> x^2 and x) -------
    auto stage = [&](int d0, int sel) {
        const int ao = sel * ATILE;
        const int bo = sel * BTILE;
        #pragma unroll
        for (int rr = 0; rr < BN; rr += 16) {
            const int r = rr + tr;
            const v4f xv = *(const v4f*)&x[(size_t)(n0 + r) * FEA + d0 + tc];
            v4bf a2, a1;
            a2[0] = (__bf16)xv[0]; a2[1] = (__bf16)xv[1];
            a2[2] = (__bf16)xv[2]; a2[3] = (__bf16)xv[3];
            a1[0] = (__bf16)(xv[0] * xv[0]); a1[1] = (__bf16)(xv[1] * xv[1]);
            a1[2] = (__bf16)(xv[2] * xv[2]); a1[3] = (__bf16)(xv[3] * xv[3]);
            *(v4bf*)&as2[ao + r * LDA + tc] = a2;
            *(v4bf*)&as1[ao + r * LDA + tc] = a1;
        }
        #pragma unroll
        for (int rr = 0; rr < BM; rr += 32) {
            const int r = rr + br;
            const size_t wb = (size_t)(m0 + r) * (2 * FEA) + d0 + bc;
            *(v4u*)&bs1[bo + r * LDA + bc] = *(const v4u*)&W[wb];
            *(v4u*)&bs2[bo + r * LDA + bc] = *(const v4u*)&W[wb + FEA];
        }
    };

    // ---- compute: 16 WMMAs against one staged BK slab ---------------------
    auto compute = [&](int sel) {
        const int ao = sel * ATILE;
        const int bo = sel * BTILE;
        #pragma unroll
        for (int kk = 0; kk < BK; kk += 32) {
            FragU fa1[2], fa2[2], fb1[2], fb2[2];
            // A 16x32 bf16 layout: lane holds row lr, K chunks
            // [lh*8, +8) and [16+lh*8, +8)
            #pragma unroll
            for (int ti = 0; ti < 2; ++ti) {
                const int o0 = ao + (wrow + ti * 16 + lr) * LDA + kk + lh * 8;
                fa1[ti].q[0] = *(const v4u*)&as1[o0];
                fa1[ti].q[1] = *(const v4u*)&as1[o0 + 16];
                fa2[ti].q[0] = *(const v4u*)&as2[o0];
                fa2[ti].q[1] = *(const v4u*)&as2[o0 + 16];
            }
            // B 32x16 bf16 layout: lane holds column lr, K chunks
            // [lh*8, +8) and [16+lh*8, +8)
            #pragma unroll
            for (int tj = 0; tj < 2; ++tj) {
                const int o = bo + (wcol + tj * 16 + lr) * LDA + kk + lh * 8;
                fb1[tj].q[0] = *(const v4u*)&bs1[o];
                fb1[tj].q[1] = *(const v4u*)&bs1[o + 16];
                fb2[tj].q[0] = *(const v4u*)&bs2[o];
                fb2[tj].q[1] = *(const v4u*)&bs2[o + 16];
            }
            #pragma unroll
            for (int ti = 0; ti < 2; ++ti)
                #pragma unroll
                for (int tj = 0; tj < 2; ++tj) {
                    acc[ti][tj] = __builtin_amdgcn_wmma_f32_16x16x32_bf16(
                        false, fa1[ti].v, false, fb1[tj].v,
                        (short)0, acc[ti][tj], false, false);
                    acc[ti][tj] = __builtin_amdgcn_wmma_f32_16x16x32_bf16(
                        false, fa2[ti].v, false, fb2[tj].v,
                        (short)0, acc[ti][tj], false, false);
                }
        }
    };

    // ---- software pipeline: stage(i+1) overlaps compute(i), 1 barrier/iter
    stage(0, 0);
    __syncthreads();
    int sel = 0;
    for (int d0 = 0; d0 < FEA; d0 += BK) {
        if (d0 + BK < FEA) stage(d0 + BK, sel ^ 1);
        compute(sel);
        __syncthreads();   // next-iter writes into `sel` are now safe
        sel ^= 1;
    }

    // ---- epilogue: C layout is VGPR v -> row (v + 8*lh), lane%16 -> col ----
    // 2000 % 16 == 0, so per-lane m-check cleanly masks padded tiles.
    // Non-temporal stores: out is write-once, keep x/W resident in L2.
    #pragma unroll
    for (int tj = 0; tj < 2; ++tj) {
        const int m = m0 + wcol + tj * 16 + lr;
        if (m >= NUM_BPE) continue;
        const float bv = bias[m];
        #pragma unroll
        for (int ti = 0; ti < 2; ++ti) {
            const int nb = n0 + wrow + ti * 16 + lh * 8;
            #pragma unroll
            for (int v = 0; v < 8; ++v)
                __builtin_nontemporal_store(acc[ti][tj][v] + bv,
                                            &out[(size_t)(nb + v) * NUM_BPE + m]);
        }
    }
}

// ---------------------------------------------------------------------------
extern "C" void kernel_launch(void* const* d_in, const int* in_sizes, int n_in,
                              void* d_out, int out_size, void* d_ws, size_t ws_size,
                              hipStream_t stream) {
    const float* x   = (const float*)d_in[0];   // (16,1024,1024) f32
    const float* mu  = (const float*)d_in[1];   // (2000,1024)    f32
    const float* cov = (const float*)d_in[2];   // (2000,1024)    f32
    float* out = (float*)d_out;                 // (16,1024,2000) f32

    // Workspace layout: W bf16 [2048][2048] then bias f32 [2048] (~8.4 MB).
    __bf16* W    = (__bf16*)d_ws;
    float*  bias = (float*)((char*)d_ws + (size_t)MPAD * (2 * FEA) * sizeof(__bf16));

    gauss_prep<<<dim3(MPAD), dim3(256), 0, stream>>>(mu, cov, W, bias);
    gauss_gemm<<<dim3(MPAD / BM, NTOT / BN), dim3(256), 0, stream>>>(x, W, bias, out);
}
